// LinearDecomMPO_65996467470409
// MI455X (gfx1250) — matrix operand order repaired and checked by reading, not compiled, not run on hardware
//
#include <hip/hip_runtime.h>
#include <stdint.h>

typedef __attribute__((ext_vector_type(16))) __bf16 v16bf;
typedef __attribute__((ext_vector_type(8)))  float  v8f;

#define M_DIM 16384
#define N_DIM 4096
#define K_DIM 4096

// ---------- fp32 -> bf16 hi/lo split helpers ----------
// RNE-rounded bf16 kept in bits [31:16]; pairs packed with one v_perm_b32.
__device__ __forceinline__ uint32_t fbits(float f){ union{float f;uint32_t u;}c; c.f=f; return c.u; }
__device__ __forceinline__ float    ubits(uint32_t u){ union{float f;uint32_t u;}c; c.u=u; return c.f; }
__device__ __forceinline__ uint32_t rne_hi(float f){
  uint32_t u = fbits(f);
  return u + 0x7FFFu + ((u >> 16) & 1u);          // bf16 value in [31:16]
}

__device__ __forceinline__ void split2(float f0, float f1, uint32_t& hi, uint32_t& lo){
  uint32_t r0 = rne_hi(f0), r1 = rne_hi(f1);
  hi = __builtin_amdgcn_perm(r1, r0, 0x07060302u); // {r1[31:16], r0[31:16]}
  float d0 = f0 - ubits(r0 & 0xFFFF0000u);
  float d1 = f1 - ubits(r1 & 0xFFFF0000u);
  uint32_t s0 = rne_hi(d0), s1 = rne_hi(d1);
  lo = __builtin_amdgcn_perm(s1, s0, 0x07060302u);
}

// ======================= MPO reconstruction chain =======================
// T1[p][r] = sum_q core0[0,a,b,q] * core1[q,c,d,r],  p=((a*8+b)*8+c)*8+d
__global__ __launch_bounds__(256) void mpo_t1(const float* __restrict__ c0,
                                              const float* __restrict__ c1,
                                              float* __restrict__ T1)
{
  int idx = blockIdx.x * 256 + threadIdx.x;      // 262144 outputs
  int r  = idx & 63;
  int p  = idx >> 6;
  int ab = p >> 6;                               // a*8+b
  int cd = p & 63;                               // c*8+d
  const float* base0 = c0 + ab * 64;             // core0[(a*8+b)*64 + q]
  const float* base1 = c1 + cd * 64 + r;         // core1[q*4096 + cd*64 + r]
  float s = 0.f;
  #pragma unroll 8
  for (int q = 0; q < 64; ++q)
    s += base0[q] * base1[(size_t)q * 4096];
  T1[idx] = s;
}

// T2[p][e][s] = sum_r T1[p][r] * core2[r,i3,o3,s],  e=i3*8+o3
__global__ __launch_bounds__(256) void mpo_t2(const float* __restrict__ T1,
                                              const float* __restrict__ c2,
                                              float* __restrict__ T2)
{
  __shared__ float t1row[64];
  int p = blockIdx.x;                            // 4096 blocks
  if (threadIdx.x < 64) t1row[threadIdx.x] = T1[p * 64 + threadIdx.x];
  __syncthreads();
  for (int oi = threadIdx.x; oi < 4096; oi += 256) {
    int e = oi >> 6, s = oi & 63;
    float acc = 0.f;
    #pragma unroll 8
    for (int r = 0; r < 64; ++r)
      acc += t1row[r] * c2[(size_t)(r * 64 + e) * 64 + s];
    T2[(size_t)p * 4096 + oi] = acc;
  }
}

// W[O][I] = sum_s T2[p][e][s] * core3[s,i4,o4]; permute to (O,I); split bf16 hi/lo.
__global__ __launch_bounds__(256) void mpo_w(const float* __restrict__ T2,
                                             const float* __restrict__ c3,
                                             uint16_t* __restrict__ Whi,
                                             uint16_t* __restrict__ Wlo)
{
  __shared__ float sc3[4096];
  for (int i = threadIdx.x; i < 4096; i += 256) sc3[i] = c3[i];
  __syncthreads();

  int gid = blockIdx.x * 256 + threadIdx.x;      // 2097152 threads
  int o4 = gid & 7;
  int e  = (gid >> 3) & 63;
  int p  = gid >> 9;

  const float* t2row = T2 + ((size_t)p * 64 + e) * 64;
  float acc[8] = {0.f,0.f,0.f,0.f,0.f,0.f,0.f,0.f};
  for (int s = 0; s < 64; ++s) {
    float t = t2row[s];
    #pragma unroll
    for (int i4 = 0; i4 < 8; ++i4)
      acc[i4] += t * sc3[s * 64 + i4 * 8 + o4];
  }

  int i1 = (p >> 9) & 7, o1 = (p >> 6) & 7, i2 = (p >> 3) & 7, o2 = p & 7;
  int i3 = e >> 3, o3 = e & 7;
  int O  = ((o1 * 8 + o2) * 8 + o3) * 8 + o4;
  int Ib = ((i1 * 8 + i2) * 8 + i3) * 8;         // + i4

  uint32_t h[4], l[4];
  #pragma unroll
  for (int j = 0; j < 4; ++j)
    split2(acc[2 * j], acc[2 * j + 1], h[j], l[j]);
  *(uint4*)(Whi + (size_t)O * 4096 + Ib) = make_uint4(h[0], h[1], h[2], h[3]);
  *(uint4*)(Wlo + (size_t)O * 4096 + Ib) = make_uint4(l[0], l[1], l[2], l[3]);
}

// ======================= main GEMM: out = x @ W^T + bias =======================
// 128x128 block tile, 8 waves (4 M x 2 N), 32x64 per wave, BK=32.
// Split-precision: acc += Ah*Bh + Ah*Bl + Al*Bh  (bf16 WMMA, f32 accumulate).
// Software pipeline: global->regs for tile k+1 issued before WMMAs of tile k;
// double-buffered LDS; one barrier per K-step.
__global__ __launch_bounds__(256) void mpo_gemm(const float*    __restrict__ x,
                                                const uint16_t* __restrict__ Whi,
                                                const uint16_t* __restrict__ Wlo,
                                                const float*    __restrict__ bias,
                                                float*          __restrict__ out)
{
  __shared__ uint32_t sAh[2][128 * 16];   // per buffer: 128 rows x 32 bf16
  __shared__ uint32_t sAl[2][128 * 16];
  __shared__ uint32_t sBh[2][128 * 16];
  __shared__ uint32_t sBl[2][128 * 16];

  const int tid  = threadIdx.x;
  const int lane = tid & 31;
  const int wave = tid >> 5;
  const int wm   = (wave & 3) * 32;    // wave M offset inside block tile
  const int wn   = (wave >> 2) * 64;   // wave N offset inside block tile
  const int l15  = lane & 15;
  const int g    = lane >> 4;

  const int nBlk = blockIdx.x * 128;
  const int mBlk = blockIdx.y * 128;

  const int sRow  = tid >> 1;          // staging row 0..127
  const int sHalf = tid & 1;           // K half: first/second 16 elements

  const float*    gA  = x   + (size_t)(mBlk + sRow) * K_DIM + sHalf * 16;
  const uint16_t* gBh = Whi + (size_t)(nBlk + sRow) * K_DIM + sHalf * 16;
  const uint16_t* gBl = Wlo + (size_t)(nBlk + sRow) * K_DIM + sHalf * 16;

  // register staging for the in-flight global tile
  float4 ra0, ra1, ra2, ra3;
  uint4  rbh0, rbh1, rbl0, rbl1;

  auto gload = [&](int k0) {
    const float4* pa = (const float4*)(gA + k0);
    ra0 = pa[0]; ra1 = pa[1]; ra2 = pa[2]; ra3 = pa[3];
    const uint4* pbh = (const uint4*)(gBh + k0);
    const uint4* pbl = (const uint4*)(gBl + k0);
    rbh0 = pbh[0]; rbh1 = pbh[1]; rbl0 = pbl[0]; rbl1 = pbl[1];
  };

  auto lstore = [&](int buf) {
    uint32_t h[8], l[8];
    split2(ra0.x, ra0.y, h[0], l[0]); split2(ra0.z, ra0.w, h[1], l[1]);
    split2(ra1.x, ra1.y, h[2], l[2]); split2(ra1.z, ra1.w, h[3], l[3]);
    split2(ra2.x, ra2.y, h[4], l[4]); split2(ra2.z, ra2.w, h[5], l[5]);
    split2(ra3.x, ra3.y, h[6], l[6]); split2(ra3.z, ra3.w, h[7], l[7]);
    uint32_t* dAh = &sAh[buf][sRow * 16 + sHalf * 8];
    ((uint4*)dAh)[0] = make_uint4(h[0], h[1], h[2], h[3]);
    ((uint4*)dAh)[1] = make_uint4(h[4], h[5], h[6], h[7]);
    uint32_t* dAl = &sAl[buf][sRow * 16 + sHalf * 8];
    ((uint4*)dAl)[0] = make_uint4(l[0], l[1], l[2], l[3]);
    ((uint4*)dAl)[1] = make_uint4(l[4], l[5], l[6], l[7]);
    uint32_t* dBh = &sBh[buf][sRow * 16 + sHalf * 8];
    ((uint4*)dBh)[0] = rbh0; ((uint4*)dBh)[1] = rbh1;
    uint32_t* dBl = &sBl[buf][sRow * 16 + sHalf * 8];
    ((uint4*)dBl)[0] = rbl0; ((uint4*)dBl)[1] = rbl1;
  };

  v8f acc[2][4] = {};

  // prologue: tile 0 into buffer 0
  gload(0);
  lstore(0);
  __syncthreads();

  const int NT = K_DIM / 32;
  for (int kt = 0; kt < NT; ++kt) {
    const int  cur  = kt & 1;
    const bool more = (kt + 1) < NT;

    // issue next tile's global loads early; waits land after the WMMA block
    if (more) gload((kt + 1) * 32);
    if (kt + 2 < NT) {                     // warm L2 two tiles ahead
      __builtin_prefetch(gA  + (kt + 2) * 32, 0, 1);
      __builtin_prefetch(gBh + (kt + 2) * 32, 0, 1);
      __builtin_prefetch(gBl + (kt + 2) * 32, 0, 1);
    }

    // ---- fragment loads (wave32 WMMA VGPR layouts) ----
    union Frag { v16bf v; uint4 q[2]; };
    Frag ah[2], al[2], bh[4], bl[4];
    #pragma unroll
    for (int mi = 0; mi < 2; ++mi) {
      int r = wm + mi * 16 + l15;
      // A 16x32 bf16: lane holds K {g*8..g*8+7} and {16+g*8..16+g*8+7}
      ah[mi].q[0] = *(const uint4*)&sAh[cur][r * 16 + g * 4];
      ah[mi].q[1] = *(const uint4*)&sAh[cur][r * 16 + 8 + g * 4];
      al[mi].q[0] = *(const uint4*)&sAl[cur][r * 16 + g * 4];
      al[mi].q[1] = *(const uint4*)&sAl[cur][r * 16 + 8 + g * 4];
    }
    #pragma unroll
    for (int ni = 0; ni < 4; ++ni) {
      int r = wn + ni * 16 + l15;
      // B 32x16 bf16: lane holds column n, K {g*8..g*8+7} and {16+g*8..16+g*8+7}
      bh[ni].q[0] = *(const uint4*)&sBh[cur][r * 16 + g * 4];
      bh[ni].q[1] = *(const uint4*)&sBh[cur][r * 16 + 8 + g * 4];
      bl[ni].q[0] = *(const uint4*)&sBl[cur][r * 16 + g * 4];
      bl[ni].q[1] = *(const uint4*)&sBl[cur][r * 16 + 8 + g * 4];
    }

    // ---- 24 WMMAs: hi*hi + hi*lo + lo*hi ----
    #pragma unroll
    for (int mi = 0; mi < 2; ++mi) {
      #pragma unroll
      for (int ni = 0; ni < 4; ++ni) {
        v8f c = acc[mi][ni];
        c = __builtin_amdgcn_wmma_f32_16x16x32_bf16(false, ah[mi].v, false, bh[ni].v, (short)0, c, false, false);
        c = __builtin_amdgcn_wmma_f32_16x16x32_bf16(false, ah[mi].v, false, bl[ni].v, (short)0, c, false, false);
        c = __builtin_amdgcn_wmma_f32_16x16x32_bf16(false, al[mi].v, false, bh[ni].v, (short)0, c, false, false);
        acc[mi][ni] = c;
      }
    }

    // ---- convert + store next tile into the other buffer ----
    if (more) lstore(cur ^ 1);
    __syncthreads();
  }

  // ---- epilogue: C layout = lane->N, VGPR r -> M = r + 8*(lane>=16) ----
  #pragma unroll
  for (int mi = 0; mi < 2; ++mi) {
    int m0 = mBlk + wm + mi * 16 + g * 8;
    #pragma unroll
    for (int ni = 0; ni < 4; ++ni) {
      int n = nBlk + wn + ni * 16 + l15;
      float bv = bias[n];
      v8f c = acc[mi][ni];
      #pragma unroll
      for (int r = 0; r < 8; ++r)
        out[(size_t)(m0 + r) * N_DIM + n] = c[r] + bv;
    }
  }
}

// ======================= host launcher =======================
extern "C" void kernel_launch(void* const* d_in, const int* in_sizes, int n_in,
                              void* d_out, int out_size, void* d_ws, size_t ws_size,
                              hipStream_t stream)
{
  (void)in_sizes; (void)n_in; (void)out_size; (void)ws_size;
  const float* x    = (const float*)d_in[0];
  const float* c0   = (const float*)d_in[1];
  const float* c1   = (const float*)d_in[2];
  const float* c2   = (const float*)d_in[3];
  const float* c3   = (const float*)d_in[4];
  const float* bias = (const float*)d_in[5];
  float* out = (float*)d_out;

  char* ws = (char*)d_ws;
  uint16_t* Whi = (uint16_t*)ws;                            // 32 MiB
  uint16_t* Wlo = (uint16_t*)(ws + (size_t)33554432);       // 32 MiB
  float*    T1  = (float*)(ws + (size_t)67108864);          // 1 MiB
  float*    T2  = (float*)(ws + (size_t)68157440);          // 64 MiB

  mpo_t1<<<1024, 256, 0, stream>>>(c0, c1, T1);
  mpo_t2<<<4096, 256, 0, stream>>>(T1, c2, T2);
  mpo_w <<<8192, 256, 0, stream>>>(T2, c3, Whi, Wlo);
  mpo_gemm<<<dim3(N_DIM / 128, M_DIM / 128), 256, 0, stream>>>(x, Whi, Wlo, bias, out);
}
